// STNBLOCK_3289944948890
// MI455X (gfx1250) — matrix operand order repaired and checked by reading, not compiled, not run on hardware
//
#include <hip/hip_runtime.h>

#define HH 128
#define WW 128
#define HWSZ (HH * WW)          // 16384
#define CC 64
#define NNB 4
#define KTOT 576                // 64 channels * 9 taps
#define NPIX (NNB * HWSZ)       // 65536

typedef __bf16 bf16_t;
typedef bf16_t v16bf __attribute__((ext_vector_type(16)));
typedef float v8f __attribute__((ext_vector_type(8)));
typedef unsigned short v16us __attribute__((ext_vector_type(16)));
typedef unsigned short v8us __attribute__((ext_vector_type(8)));

struct AFragBits { v8us lo, hi; };   // 32 bytes, bit_cast -> v16bf

// Native cast -> hardware v_cvt_(pk_)bf16_f32 instead of manual RNE bit math.
__device__ __forceinline__ unsigned short f32_bf16(float f) {
    return __builtin_bit_cast(unsigned short, (__bf16)f);
}

// ---------------------------------------------------------------------------
// Kernel 1: offset prediction. t = conv3x3(x, wt) + bt (4 channels), then
// py = y + t0*ky + t1*kx ; px = x + t2*ky + t3*kx   (base grid cancels).
// coords layout: [pixel][tap][2] floats.
// ---------------------------------------------------------------------------
__global__ __launch_bounds__(256)
void offsets_kernel(const float* __restrict__ x, const float* __restrict__ wt,
                    const float* __restrict__ bt, float* __restrict__ coords) {
    int gm = blockIdx.x * 256 + threadIdx.x;
    if (gm >= NPIX) return;
    int n = gm >> 14;
    int rem = gm & (HWSZ - 1);
    int y = rem >> 7;
    int xc = rem & (WW - 1);

    float t0 = bt[0], t1 = bt[1], t2 = bt[2], t3 = bt[3];
    for (int c = 0; c < CC; ++c) {
        const float* plane = x + (size_t)(n * CC + c) * HWSZ;
        float v[9];
#pragma unroll
        for (int k = 0; k < 9; ++k) {
            int dy = k / 3 - 1, dx = k % 3 - 1;
            int yy = y + dy, xx = xc + dx;
            v[k] = (yy >= 0 && yy < HH && xx >= 0 && xx < WW) ? plane[yy * WW + xx] : 0.f;
        }
        const float* w0 = wt + c * 9;   // wt: (4, 64, 3, 3) -> tc*576 + c*9 + k
#pragma unroll
        for (int k = 0; k < 9; ++k) {
            float vk = v[k];
            t0 = fmaf(vk, w0[k], t0);
            t1 = fmaf(vk, w0[576 + k], t1);
            t2 = fmaf(vk, w0[1152 + k], t2);
            t3 = fmaf(vk, w0[1728 + k], t3);
        }
    }
    float* cp = coords + (size_t)gm * 18;
#pragma unroll
    for (int k = 0; k < 9; ++k) {
        float ky = (float)(k / 3 - 1), kx = (float)(k % 3 - 1);
        cp[2 * k]     = (float)y  + t0 * ky + t1 * kx;
        cp[2 * k + 1] = (float)xc + t2 * ky + t3 * kx;
    }
}

// ---------------------------------------------------------------------------
// Kernel 2: deformable conv = bilinear-gather + bf16 WMMA GEMM.
// Workgroup: 32 pixels (M) x 64 outputs (N). 8 waves, one 16x16 tile each.
// K = c*9 + tap; 18 double-buffered chunks of 32, one barrier per chunk.
// LDS A tile: [m][kk] (32x32 bf16). LDS B tile: [o][kk] (64x32 bf16,
// o-major so a B fragment is one contiguous 32-byte run per lane).
// ---------------------------------------------------------------------------
__global__ __launch_bounds__(256)
void deform_wmma_kernel(const float* __restrict__ src,
                        const float* __restrict__ coords,
                        const float* __restrict__ wd,
                        float* __restrict__ out) {
    __shared__ alignas(16) unsigned short sA[2][32 * 32];   // [buf][m*32+kk]
    __shared__ alignas(16) unsigned short sB[2][64 * 32];   // [buf][o*32+kk]
    __shared__ float sPy[32 * 9];
    __shared__ float sPx[32 * 9];

    int tid = threadIdx.x;
    int wgStart = blockIdx.x * 32;

    for (int i = tid; i < 32 * 9; i += 256) {
        int m = i / 9, tap = i - m * 9;
        const float* cp = coords + (size_t)(wgStart + m) * 18 + tap * 2;
        sPy[i] = cp[0];
        sPx[i] = cp[1];
    }
    __builtin_prefetch(wd, 0, 3);   // global_prefetch_b8

    // -------- staging: fill one K-chunk (A gather + B weights) ------------
    auto stage = [&](int kc, unsigned short* bA, unsigned short* bB) {
        unsigned int* bB32 = (unsigned int*)bB;
        unsigned int* bA32 = (unsigned int*)bA;
        // B: 64 o x 32 kk = 1024 packed-pair stores
#pragma unroll
        for (int i = 0; i < 4; ++i) {
            int uidx = tid + i * 256;            // 0..1023
            int o = uidx >> 4, kp = uidx & 15;   // kk = 2*kp
            const float* wp = wd + (size_t)o * KTOT + kc * 32 + 2 * kp;
            unsigned lo = f32_bf16(wp[0]);
            unsigned hi = f32_bf16(wp[1]);
            bB32[uidx] = lo | (hi << 16);
        }
        // A: 32 m x 32 kk = 512 packed-pair stores (bilinear gather)
#pragma unroll
        for (int i = 0; i < 2; ++i) {
            int uidx = tid + i * 256;            // 0..511
            int m = uidx >> 4, kp = uidx & 15;
            int gm = wgStart + m;
            int n = gm >> 14;
            unsigned pk = 0;
#pragma unroll
            for (int h = 0; h < 2; ++h) {
                int K = kc * 32 + 2 * kp + h;
                int c = K / 9, tap = K - c * 9;
                const float* plane = src + (size_t)(n * CC + c) * HWSZ;
                float py = sPy[m * 9 + tap];
                float px = sPx[m * 9 + tap];
                float y0f = floorf(py), x0f = floorf(px);
                float wy = py - y0f, wx = px - x0f;
                int iy0 = (int)y0f, ix0 = (int)x0f;
                int iy1 = iy0 + 1, ix1 = ix0 + 1;
                bool vy0 = (iy0 >= 0) & (iy0 < HH);
                bool vy1 = (iy1 >= 0) & (iy1 < HH);
                bool vx0 = (ix0 >= 0) & (ix0 < WW);
                bool vx1 = (ix1 >= 0) & (ix1 < WW);
                float v00 = (vy0 & vx0) ? plane[iy0 * WW + ix0] : 0.f;
                float v01 = (vy0 & vx1) ? plane[iy0 * WW + ix1] : 0.f;
                float v10 = (vy1 & vx0) ? plane[iy1 * WW + ix0] : 0.f;
                float v11 = (vy1 & vx1) ? plane[iy1 * WW + ix1] : 0.f;
                float val = v00 * (1.f - wy) * (1.f - wx) + v01 * (1.f - wy) * wx
                          + v10 * wy * (1.f - wx)         + v11 * wy * wx;
                pk |= ((unsigned)f32_bf16(val)) << (16 * h);
            }
            bA32[uidx] = pk;
        }
    };

    int lane = tid & 31;
    int wv = tid >> 5;      // 0..7
    int mt = wv >> 2;       // 0..1  (16-pixel half)
    int nt = wv & 3;        // 0..3  (16-output column block)
    int half = lane >> 4;
    int l16 = lane & 15;

    v8f acc = {};
    stage(0, sA[0], sB[0]);
    int p = 0;

    for (int kc = 0; kc < 18; ++kc) {
        __syncthreads();    // buf[p] staged; buf[p^1] fully consumed

        // ---- fragment loads (wide LDS reads) ----
        // A: lane row = mt*16+l16; elems e -> kk=half*16+e
        const unsigned short* aRow = sA[p] + (mt * 16 + l16) * 32;
        AFragBits af;
        af.lo = *(const v8us*)(aRow + half * 16);       // 16B aligned
        af.hi = *(const v8us*)(aRow + half * 16 + 8);   // 16B aligned
        v16bf a = __builtin_bit_cast(v16bf, af);
        // B: col = nt*16+l16, K = half*16 + e  -> 16 contiguous shorts
        const unsigned short* bRow = sB[p] + (nt * 16 + l16) * 32 + half * 16;
        v16us bu = *(const v16us*)bRow;                 // 32B aligned
        v16bf b = __builtin_bit_cast(v16bf, bu);

        // ---- overlap: stage next chunk into the other buffer ----
        if (kc < 17) stage(kc + 1, sA[p ^ 1], sB[p ^ 1]);

        acc = __builtin_amdgcn_wmma_f32_16x16x32_bf16(false, a, false, b,
                                                      (short)0, acc, false, false);
        p ^= 1;
    }

    // D layout: VGPR r, lane L: M = r + 8*(L/16), Ncol = L%16
#pragma unroll
    for (int r = 0; r < 8; ++r) {
        int mlocal = mt * 16 + r + 8 * half;
        int o = nt * 16 + l16;
        int gm = wgStart + mlocal;
        int n = gm >> 14, rem = gm & (HWSZ - 1);
        out[(size_t)(n * CC + o) * HWSZ + rem] = acc[r];
    }
}

// ---------------------------------------------------------------------------
// Kernel 3: per-channel batch statistics (mean, rstd) over N,H,W.
// ---------------------------------------------------------------------------
__global__ __launch_bounds__(256)
void bnstats_kernel(const float* __restrict__ src, float* __restrict__ stats) {
    __shared__ float ssum[256];
    __shared__ float ssq[256];
    int c = blockIdx.x, tid = threadIdx.x;
    float s = 0.f, q = 0.f;
    for (int n = 0; n < NNB; ++n) {
        const float* p = src + (size_t)(n * CC + c) * HWSZ;
        for (int i = tid; i < HWSZ; i += 256) {
            float v = p[i];
            s += v;
            q = fmaf(v, v, q);
        }
    }
    ssum[tid] = s; ssq[tid] = q;
    __syncthreads();
    for (int st = 128; st > 0; st >>= 1) {
        if (tid < st) { ssum[tid] += ssum[tid + st]; ssq[tid] += ssq[tid + st]; }
        __syncthreads();
    }
    if (tid == 0) {
        const float inv = 1.f / (float)(NNB * HWSZ);
        float m = ssum[0] * inv;
        float var = ssq[0] * inv - m * m;
        stats[2 * c] = m;
        stats[2 * c + 1] = rsqrtf(var + 1e-5f);
    }
}

__global__ __launch_bounds__(256)
void bnapply_relu_kernel(const float* __restrict__ src, const float* __restrict__ stats,
                         const float* __restrict__ g, const float* __restrict__ b,
                         float* __restrict__ dst) {
    int idx = blockIdx.x * 256 + threadIdx.x;
    int c = (idx >> 14) & 63;
    float v = (src[idx] - stats[2 * c]) * stats[2 * c + 1] * g[c] + b[c];
    dst[idx] = fmaxf(v, 0.f);
}

__global__ __launch_bounds__(256)
void bnapply_res_relu_kernel(const float* __restrict__ src, const float* __restrict__ stats,
                             const float* __restrict__ g, const float* __restrict__ b,
                             const float* __restrict__ resid, float* __restrict__ dst) {
    int idx = blockIdx.x * 256 + threadIdx.x;
    int c = (idx >> 14) & 63;
    float v = (src[idx] - stats[2 * c]) * stats[2 * c + 1] * g[c] + b[c] + resid[idx];
    dst[idx] = fmaxf(v, 0.f);
}

// ---------------------------------------------------------------------------
extern "C" void kernel_launch(void* const* d_in, const int* in_sizes, int n_in,
                              void* d_out, int out_size, void* d_ws, size_t ws_size,
                              hipStream_t stream) {
    const float* x   = (const float*)d_in[0];
    const float* wt1 = (const float*)d_in[1];
    const float* bt1 = (const float*)d_in[2];
    const float* wd1 = (const float*)d_in[3];
    const float* g1  = (const float*)d_in[4];
    const float* b1  = (const float*)d_in[5];
    const float* wt2 = (const float*)d_in[6];
    const float* bt2 = (const float*)d_in[7];
    const float* wd2 = (const float*)d_in[8];
    const float* g2  = (const float*)d_in[9];
    const float* b2  = (const float*)d_in[10];
    float* out = (float*)d_out;

    float* ws = (float*)d_ws;
    float* coords1 = ws;                      // 65536*18 = 1,179,648
    float* coords2 = ws + 1179648;            // 1,179,648
    float* ybuf    = ws + 2359296;            // 4,194,304 (reused for y1 and y2)
    float* hbuf    = ws + 6553600;            // 4,194,304
    float* stats1  = ws + 10747904;           // 128
    float* stats2  = ws + 10748032;           // 128

    offsets_kernel<<<NPIX / 256, 256, 0, stream>>>(x, wt1, bt1, coords1);
    offsets_kernel<<<NPIX / 256, 256, 0, stream>>>(x, wt2, bt2, coords2);

    deform_wmma_kernel<<<NPIX / 32, 256, 0, stream>>>(x, coords1, wd1, ybuf);
    bnstats_kernel<<<CC, 256, 0, stream>>>(ybuf, stats1);
    bnapply_relu_kernel<<<(NNB * CC * HWSZ) / 256, 256, 0, stream>>>(ybuf, stats1, g1, b1, hbuf);

    deform_wmma_kernel<<<NPIX / 32, 256, 0, stream>>>(hbuf, coords2, wd2, ybuf);
    bnstats_kernel<<<CC, 256, 0, stream>>>(ybuf, stats2);
    bnapply_res_relu_kernel<<<(NNB * CC * HWSZ) / 256, 256, 0, stream>>>(ybuf, stats2, g2, b2, x, out);
}
